// MultiHeadSelfAttention_91173565759633
// MI455X (gfx1250) — matrix operand order repaired and checked by reading, not compiled, not run on hardware
//
#include <hip/hip_runtime.h>
#include <hip/hip_bf16.h>
#include <cstdint>

// ---------- vector types matching CDNA5 WMMA operand shapes ----------
typedef __attribute__((ext_vector_type(16))) __bf16 v16bf;
typedef __attribute__((ext_vector_type(8)))  __bf16 v8bf;
typedef __attribute__((ext_vector_type(4)))  __bf16 v4bf;
typedef __attribute__((ext_vector_type(8)))  float  v8f;
typedef __attribute__((ext_vector_type(4)))  int    v4i;

#define AS1 __attribute__((address_space(1)))
#define AS3 __attribute__((address_space(3)))

// Async global->LDS copy path (CDNA5). Fall back to sync double-buffering if the
// builtins are not available in this toolchain.
#if defined(__has_builtin)
#if __has_builtin(__builtin_amdgcn_global_load_async_to_lds_b128) && \
    __has_builtin(__builtin_amdgcn_s_wait_asynccnt)
#define USE_ASYNC 1
#endif
#if __has_builtin(__builtin_amdgcn_mov_dpp8) && __has_builtin(__builtin_amdgcn_update_dpp)
#define USE_DPP 1
#endif
#endif
#ifndef USE_ASYNC
#define USE_ASYNC 0
#endif
#ifndef USE_DPP
#define USE_DPP 0
#endif

static __device__ __forceinline__ v16bf cat16(v8bf lo, v8bf hi) {
  return __builtin_shufflevector(lo, hi, 0,1,2,3,4,5,6,7,8,9,10,11,12,13,14,15);
}
static __device__ __forceinline__ v8f wmma_bf16f32(v16bf a, v16bf b, v8f c) {
  // D(f32 16x16) = A(bf16 16x32) * B(bf16 32x16) + C
  return __builtin_amdgcn_wmma_f32_16x16x32_bf16(false, a, false, b, (short)0, c, false, false);
}
static __device__ __forceinline__ v8f zero8() {
  v8f z;
#pragma unroll
  for (int i = 0; i < 8; ++i) z[i] = 0.f;
  return z;
}
#if USE_ASYNC
static __device__ __forceinline__ void async_cp16(const __bf16* g, __bf16* l) {
  // signature (from hipcc diagnostic): (global int4*, lds int4*, imm offset, imm cpol)
  __builtin_amdgcn_global_load_async_to_lds_b128((AS1 v4i*)g, (AS3 v4i*)l, 0, 0);
}
#endif

// Max across each 16-lane half (lanes 0-15 and 16-31 independent).
#if USE_DPP
constexpr int dpp8_xor_sel(int m) {
  int s = 0;
  for (int j = 0; j < 8; ++j) s |= ((j ^ m) & 7) << (3 * j);
  return s;
}
static __device__ __forceinline__ float halfwave_max16(float x) {
  union fi { float f; int i; };
  fi u, v;
  u.f = x;
  // xor-1/2/4 inside each 8-lane DPP8 group (stays within the 16-lane half)
  v.i = __builtin_amdgcn_mov_dpp8(u.i, dpp8_xor_sel(1)); u.f = fmaxf(u.f, v.f);
  v.i = __builtin_amdgcn_mov_dpp8(u.i, dpp8_xor_sel(2)); u.f = fmaxf(u.f, v.f);
  v.i = __builtin_amdgcn_mov_dpp8(u.i, dpp8_xor_sel(4)); u.f = fmaxf(u.f, v.f);
  // i <-> i+-8 pairing inside the 16-lane DPP row: row_ror:8 (0x128)
  v.i = __builtin_amdgcn_update_dpp(u.i, u.i, 0x128, 0xf, 0xf, true);
  u.f = fmaxf(u.f, v.f);
  return u.f;
}
#else
static __device__ __forceinline__ float halfwave_max16(float x) {
#pragma unroll
  for (int mk = 1; mk < 16; mk <<= 1) x = fmaxf(x, __shfl_xor(x, mk, 32));
  return x;
}
#endif

// ---------------- fp32 -> bf16 conversion (vectorized) ----------------
__global__ __launch_bounds__(256)
void cvt_f32_to_bf16(const float* __restrict__ in, __bf16* __restrict__ out, int n4) {
  int i = blockIdx.x * 256 + threadIdx.x;
  if (i >= n4) return;
  float4 v = ((const float4*)in)[i];
  v4bf o;
  o[0] = (__bf16)v.x; o[1] = (__bf16)v.y; o[2] = (__bf16)v.z; o[3] = (__bf16)v.w;
  ((v4bf*)out)[i] = o;
}

// ---------------- bf16 GEMM: out[m][n] = A[m][:] . Bw[n][:] + bias[n] ----------------
// A: [M][K] bf16 row-major, Bw: [N][K] bf16 row-major (torch Linear weight layout).
// Block tile 128(M) x 128(N), K-step 32, double-buffered LDS (async copies on CDNA5).
// 256 threads = 8 waves (4 in M x 2 in N), wave tile 32x64 -> 8 WMMA per K-step.
// QKV_EPI=1: scatter bf16 into [3][B][H][N][D] with V transposed to [B][H][D][N];
// QKV_EPI=0: fp32 row-major [M][N].
template <int QKV_EPI>
__global__ __launch_bounds__(256)
void gemm_bf16_kernel(const __bf16* __restrict__ A, const __bf16* __restrict__ Bw,
                      const float* __restrict__ bias, void* __restrict__ outp,
                      int M, int N, int K) {
  constexpr int LDA = 40;  // 32 + 8 pad: 16B-aligned rows, conflict-free (20 banks/row)
  constexpr int LDB = 40;
  __shared__ __bf16 As[2][128 * LDA];
  __shared__ __bf16 Bs[2][128 * LDB];

  const int tid  = threadIdx.x;
  const int lane = tid & 31, wave = tid >> 5;
  const int half = lane >> 4, l16 = lane & 15;
  const int wm = wave >> 1, wn = wave & 1;  // 4 waves in M x 2 waves in N
  const int m0 = blockIdx.y * 128;
  const int n0 = blockIdx.x * 128;

  v8f acc[2][4];
#pragma unroll
  for (int r = 0; r < 2; ++r)
#pragma unroll
    for (int f = 0; f < 4; ++f) acc[r][f] = zero8();

  const int nIter = K / 32;

#if USE_ASYNC
  // ---- prologue: async-fill buffer 0 (4 async ops per wave) ----
#pragma unroll
  for (int it = 0; it < 2; ++it) {
    int idx = tid + it * 256, row = idx >> 2, seg = idx & 3;
    async_cp16(A  + (size_t)(m0 + row) * K + seg * 8, &As[0][row * LDA + seg * 8]);
    async_cp16(Bw + (size_t)(n0 + row) * K + seg * 8, &Bs[0][row * LDB + seg * 8]);
  }
#else
#pragma unroll
  for (int it = 0; it < 2; ++it) {
    int idx = tid + it * 256, row = idx >> 2, seg = idx & 3;
    *(v8bf*)(&As[0][row * LDA + seg * 8]) = *(const v8bf*)(A  + (size_t)(m0 + row) * K + seg * 8);
    *(v8bf*)(&Bs[0][row * LDB + seg * 8]) = *(const v8bf*)(Bw + (size_t)(n0 + row) * K + seg * 8);
  }
  v8bf pa[2], pb[2];
#endif

  for (int i = 0; i < nIter; ++i) {
    const int cur = i & 1;
#if USE_ASYNC
    if (i + 1 < nIter) {  // issue next tile into the other buffer, then drain previous
#pragma unroll
      for (int it = 0; it < 2; ++it) {
        int idx = tid + it * 256, row = idx >> 2, seg = idx & 3;
        async_cp16(A  + (size_t)(m0 + row) * K + (i + 1) * 32 + seg * 8,
                   &As[cur ^ 1][row * LDA + seg * 8]);
        async_cp16(Bw + (size_t)(n0 + row) * K + (i + 1) * 32 + seg * 8,
                   &Bs[cur ^ 1][row * LDB + seg * 8]);
      }
      __builtin_amdgcn_s_wait_asynccnt(4);  // in-order: <=4 pending => previous tile done
    } else {
      __builtin_amdgcn_s_wait_asynccnt(0);
    }
    __syncthreads();
#else
    if (i + 1 < nIter) {  // register prefetch of the next tile
#pragma unroll
      for (int it = 0; it < 2; ++it) {
        int idx = tid + it * 256, row = idx >> 2, seg = idx & 3;
        pa[it] = *(const v8bf*)(A  + (size_t)(m0 + row) * K + (i + 1) * 32 + seg * 8);
        pb[it] = *(const v8bf*)(Bw + (size_t)(n0 + row) * K + (i + 1) * 32 + seg * 8);
      }
    }
    __syncthreads();
#endif

    // ---- compute on buffer `cur` ----
    const __bf16* as = As[cur];
    const __bf16* bs = Bs[cur];
    v16bf af[2];
#pragma unroll
    for (int r = 0; r < 2; ++r) {
      // A frag: row = wm*32 + r*16 + l16; K = {8*half+0..7, 8*half+16..23}
      const __bf16* ap = as + (wm * 32 + r * 16 + l16) * LDA + half * 8;
      af[r] = cat16(*(const v8bf*)ap, *(const v8bf*)(ap + 16));
    }
#pragma unroll
    for (int f = 0; f < 4; ++f) {
      // B frag: col n = wn*64 + f*16 + l16; K = {8*half+0..7, 8*half+16..23}
      const __bf16* bp = bs + (wn * 64 + f * 16 + l16) * LDB + half * 8;
      v16bf bf = cat16(*(const v8bf*)bp, *(const v8bf*)(bp + 16));
#pragma unroll
      for (int r = 0; r < 2; ++r) acc[r][f] = wmma_bf16f32(af[r], bf, acc[r][f]);
    }

#if USE_ASYNC
    __syncthreads();  // all reads of `cur` done before iter i+1 refills it
#else
    if (i + 1 < nIter) {
      __syncthreads();
#pragma unroll
      for (int it = 0; it < 2; ++it) {
        int idx = tid + it * 256, row = idx >> 2, seg = idx & 3;
        *(v8bf*)(&As[cur ^ 1][row * LDA + seg * 8]) = pa[it];
        *(v8bf*)(&Bs[cur ^ 1][row * LDB + seg * 8]) = pb[it];
      }
    }
#endif
  }

  // Epilogue. C frag: element i -> row m = i + 8*half, col n = l16 (per 16x16 frag).
#pragma unroll
  for (int r = 0; r < 2; ++r)
#pragma unroll
    for (int f = 0; f < 4; ++f) {
      int n = n0 + wn * 64 + f * 16 + l16;
      float bv = bias[n];
#pragma unroll
      for (int i = 0; i < 8; ++i) {
        int m = m0 + wm * 32 + r * 16 + i + 8 * half;
        float v = acc[r][f][i] + bv;
        if constexpr (QKV_EPI) {
          int s = n / 768, rr = n - s * 768;
          int hh = rr >> 6, d = rr & 63;
          int bb = m >> 11, tok = m & 2047;
          size_t off;
          if (s == 2)  // V stored transposed: [B][H][D][N]
            off = ((((size_t)4 + bb) * 12 + hh) * 64 + d) * 2048 + tok;
          else
            off = ((((size_t)s * 2 + bb) * 12 + hh) * 2048 + tok) * 64 + d;
          ((__bf16*)outp)[off] = (__bf16)v;
        } else {
          ((float*)outp)[(size_t)m * N + n] = v;
        }
      }
    }
}

// ---------------- Flash attention ----------------
// grid = (N/128, B*H). 8 waves, each wave owns 16 query rows end-to-end.
// Streams 64-key blocks with double-buffered (async on CDNA5) K and V^T tiles:
// S = Q K^T (bf16 WMMA, f32 acc), online softmax in registers (base-2 domain),
// row max via DPP butterflies, row sums via P @ ones on the matrix unit,
// P re-layouts C-frag -> A-frag through per-wave LDS, then P V accumulate.
__global__ __launch_bounds__(256)
void flash_attn_kernel(const __bf16* __restrict__ qkv, __bf16* __restrict__ obuf) {
  constexpr int LDK = 72;  // 64 + 8 pad: 16B-aligned rows, conflict-free
  __shared__ __bf16 Ks[2][64 * LDK];    // K tile  [key][d]
  __shared__ __bf16 Vts[2][64 * LDK];   // V tile  [d][key] (pre-transposed in global)
  __shared__ __bf16 Ps[8 * 16 * LDK];   // per-wave P patch [16 rows][64 keys]

  const int tid  = threadIdx.x;
  const int lane = tid & 31, wave = tid >> 5;
  const int half = lane >> 4, l16 = lane & 15;
  const int bh = blockIdx.y;
  const int b = bh / 12, h = bh % 12;
  const size_t SH = (size_t)2048 * 64;
  const __bf16* qb  = qkv + (((size_t)0 * 2 + b) * 12 + h) * SH;
  const __bf16* kbp = qkv + (((size_t)1 * 2 + b) * 12 + h) * SH;             // [n][d]
  const __bf16* vtp = qkv + ((((size_t)4 + b) * 12 + h) * 64) * 2048;        // [d][n]

  const int qrow0 = blockIdx.x * 128 + wave * 16;

  // Q fragments for D=64 (two K-steps of 32), kept in registers for the whole kernel.
  v16bf qf[2];
  {
    const __bf16* qp = qb + (size_t)(qrow0 + l16) * 64 + half * 8;
#pragma unroll
    for (int s = 0; s < 2; ++s)
      qf[s] = cat16(*(const v8bf*)(qp + s * 32), *(const v8bf*)(qp + s * 32 + 16));
  }

  // all-ones B fragment: rowsum(P) = P @ ones lands replicated in every column,
  // matching the [i + 8*half] row-stats layout exactly.
  v16bf ones;
#pragma unroll
  for (int i = 0; i < 16; ++i) ones[i] = (__bf16)1.0f;

  float mrow[8], lrow[8];
  v8f of[4];
#pragma unroll
  for (int i = 0; i < 8; ++i) { mrow[i] = -3.0e38f; lrow[i] = 0.f; }
#pragma unroll
  for (int f = 0; f < 4; ++f) of[f] = zero8();

  // softmax computed in base-2 domain: exp2(c*s - max(c*s)) == exp(scale*(s - smax))
  const float SCL2 = 0.125f * 1.44269504088896f;  // head_dim^-0.5 * log2(e)

#if USE_ASYNC
#pragma unroll
  for (int it = 0; it < 2; ++it) {
    int idx = tid + it * 256, row = idx >> 3, seg = idx & 7;
    async_cp16(kbp + (size_t)row * 64 + seg * 8,   &Ks[0][row * LDK + seg * 8]);
    async_cp16(vtp + (size_t)row * 2048 + seg * 8, &Vts[0][row * LDK + seg * 8]);
  }
#else
#pragma unroll
  for (int it = 0; it < 2; ++it) {
    int idx = tid + it * 256, row = idx >> 3, seg = idx & 7;
    *(v8bf*)(&Ks[0][row * LDK + seg * 8])  = *(const v8bf*)(kbp + (size_t)row * 64 + seg * 8);
    *(v8bf*)(&Vts[0][row * LDK + seg * 8]) = *(const v8bf*)(vtp + (size_t)row * 2048 + seg * 8);
  }
  v8bf pk[2], pv[2];
#endif

  for (int kb = 0; kb < 32; ++kb) {
    const int cur = kb & 1;
#if USE_ASYNC
    if (kb + 1 < 32) {
#pragma unroll
      for (int it = 0; it < 2; ++it) {
        int idx = tid + it * 256, row = idx >> 3, seg = idx & 7;
        async_cp16(kbp + (size_t)((kb + 1) * 64 + row) * 64 + seg * 8,
                   &Ks[cur ^ 1][row * LDK + seg * 8]);
        async_cp16(vtp + (size_t)row * 2048 + (kb + 1) * 64 + seg * 8,
                   &Vts[cur ^ 1][row * LDK + seg * 8]);
      }
      __builtin_amdgcn_s_wait_asynccnt(4);
    } else {
      __builtin_amdgcn_s_wait_asynccnt(0);
    }
    __syncthreads();
#else
    if (kb + 1 < 32) {
#pragma unroll
      for (int it = 0; it < 2; ++it) {
        int idx = tid + it * 256, row = idx >> 3, seg = idx & 7;
        pk[it] = *(const v8bf*)(kbp + (size_t)((kb + 1) * 64 + row) * 64 + seg * 8);
        pv[it] = *(const v8bf*)(vtp + (size_t)row * 2048 + (kb + 1) * 64 + seg * 8);
      }
    }
    __syncthreads();
#endif

    // ---- S = Q * K^T : 4 frags of 16x16 keys, K-dim = d = 64 ----
    v8f sf[4];
#pragma unroll
    for (int f = 0; f < 4; ++f) {
      v8f c = zero8();
#pragma unroll
      for (int s = 0; s < 2; ++s) {
        const __bf16* kp = &Ks[cur][(f * 16 + l16) * LDK + s * 32 + half * 8];
        c = wmma_bf16f32(qf[s], cat16(*(const v8bf*)kp, *(const v8bf*)(kp + 16)), c);
      }
      sf[f] = c;
    }

    // ---- scale (base-2 domain) + row max (stats replicated per 16-lane half) ----
    float rmax[8];
#pragma unroll
    for (int i = 0; i < 8; ++i) rmax[i] = -3.0e38f;
#pragma unroll
    for (int f = 0; f < 4; ++f)
#pragma unroll
      for (int i = 0; i < 8; ++i) { sf[f][i] *= SCL2; rmax[i] = fmaxf(rmax[i], sf[f][i]); }
#pragma unroll
    for (int i = 0; i < 8; ++i) rmax[i] = halfwave_max16(rmax[i]);

    float alpha[8];
#pragma unroll
    for (int i = 0; i < 8; ++i) {
      float mn = fmaxf(mrow[i], rmax[i]);
      alpha[i] = __builtin_amdgcn_exp2f(mrow[i] - mn);
      mrow[i] = mn;
    }
#pragma unroll
    for (int f = 0; f < 4; ++f)
#pragma unroll
      for (int i = 0; i < 8; ++i)
        sf[f][i] = __builtin_amdgcn_exp2f(sf[f][i] - mrow[i]);
#pragma unroll
    for (int f = 0; f < 4; ++f)
#pragma unroll
      for (int i = 0; i < 8; ++i) of[f][i] *= alpha[i];

    // ---- P: C-layout f32 -> per-wave LDS -> A-layout bf16 (same-wave DS in-order) ----
    __bf16* pw = Ps + wave * 16 * LDK;
#pragma unroll
    for (int f = 0; f < 4; ++f)
#pragma unroll
      for (int i = 0; i < 8; ++i)
        pw[(i + 8 * half) * LDK + f * 16 + l16] = (__bf16)sf[f][i];

    v16bf pf[2];
#pragma unroll
    for (int s = 0; s < 2; ++s) {
      const __bf16* pp = pw + l16 * LDK + s * 32 + half * 8;
      pf[s] = cat16(*(const v8bf*)pp, *(const v8bf*)(pp + 16));
    }

    // ---- row sums on the matrix unit: rs = P @ ones (no cross-lane shuffles) ----
    v8f rs = zero8();
#pragma unroll
    for (int s = 0; s < 2; ++s) rs = wmma_bf16f32(pf[s], ones, rs);
#pragma unroll
    for (int i = 0; i < 8; ++i) lrow[i] = lrow[i] * alpha[i] + rs[i];

    // ---- O += P * V : K-dim = 64 keys (two K-steps) ----
#pragma unroll
    for (int s = 0; s < 2; ++s)
#pragma unroll
      for (int f = 0; f < 4; ++f) {
        const __bf16* vp = &Vts[cur][(f * 16 + l16) * LDK + s * 32 + half * 8];
        of[f] = wmma_bf16f32(pf[s], cat16(*(const v8bf*)vp, *(const v8bf*)(vp + 16)), of[f]);
      }

#if USE_ASYNC
    __syncthreads();  // all reads of `cur` done before iter kb+1 refills it
#else
    if (kb + 1 < 32) {
      __syncthreads();
#pragma unroll
      for (int it = 0; it < 2; ++it) {
        int idx = tid + it * 256, row = idx >> 3, seg = idx & 7;
        *(v8bf*)(&Ks[cur ^ 1][row * LDK + seg * 8])  = pk[it];
        *(v8bf*)(&Vts[cur ^ 1][row * LDK + seg * 8]) = pv[it];
      }
    }
#endif
  }

  // Normalize and store O as bf16 into [B*N][C] for the projection GEMM.
#pragma unroll
  for (int i = 0; i < 8; ++i) lrow[i] = 1.f / lrow[i];
#pragma unroll
  for (int f = 0; f < 4; ++f)
#pragma unroll
    for (int i = 0; i < 8; ++i) {
      int d = f * 16 + l16;
      int tok = qrow0 + i + 8 * half;
      obuf[((size_t)b * 2048 + tok) * 768 + h * 64 + d] = (__bf16)(of[f][i] * lrow[i]);
    }
}

// ---------------- launcher ----------------
extern "C" void kernel_launch(void* const* d_in, const int* in_sizes, int n_in,
                              void* d_out, int out_size, void* d_ws, size_t ws_size,
                              hipStream_t stream) {
  const float* x      = (const float*)d_in[0];
  const float* w_qkv  = (const float*)d_in[1];
  const float* b_qkv  = (const float*)d_in[2];
  const float* w_proj = (const float*)d_in[3];
  const float* b_proj = (const float*)d_in[4];

  char* ws = (char*)d_ws;
  __bf16* xb     = (__bf16*)(ws);                // 4096*768*2        = 6,291,456 B
  __bf16* wqkvb  = (__bf16*)(ws + 6291456);      // 2304*768*2       = 3,538,944 B
  __bf16* wprojb = (__bf16*)(ws + 9830400);      // 768*768*2        = 1,179,648 B
  __bf16* qkvb   = (__bf16*)(ws + 11010048);     // 3*2*12*2048*64*2 = 18,874,368 B
  __bf16* obuf   = (__bf16*)(ws + 29884416);     // 4096*768*2       = 6,291,456 B
  // total workspace: 36,175,872 B

  cvt_f32_to_bf16<<<3072, 256, 0, stream>>>(x, xb, 786432);
  cvt_f32_to_bf16<<<1728, 256, 0, stream>>>(w_qkv, wqkvb, 442368);
  cvt_f32_to_bf16<<<576, 256, 0, stream>>>(w_proj, wprojb, 147456);

  // QKV projection: [4096 x 768] x [2304 x 768]^T -> bf16 Q,K:[B][H][N][D], V:[B][H][D][N]
  gemm_bf16_kernel<1><<<dim3(18, 32), 256, 0, stream>>>(xb, wqkvb, b_qkv, qkvb,
                                                        4096, 2304, 768);
  // Flash attention over 24 (b,h) pairs, 16 row-blocks each
  flash_attn_kernel<<<dim3(16, 24), 256, 0, stream>>>(qkvb, obuf);
  // Output projection: [4096 x 768] x [768 x 768]^T -> fp32 d_out
  gemm_bf16_kernel<0><<<dim3(6, 32), 256, 0, stream>>>(obuf, wprojb, b_proj, d_out,
                                                       4096, 768, 768);
}
